// LlamaModel_80917183856893
// MI455X (gfx1250) — compile-verified, matrix operand order checked by reading, not run on hardware
//
#include <hip/hip_runtime.h>
#include <hip/hip_bf16.h>

#define S_LEN 2048
#define DIM   1024
#define NH    16
#define NKV   8
#define HD    64
#define HID   2816
#define NL    2
#define VOCAB 32000

typedef __attribute__((ext_vector_type(16))) _Float16 v16h;
typedef __attribute__((ext_vector_type(8)))  _Float16 v8h;
typedef __attribute__((ext_vector_type(4)))  _Float16 v4h;
typedef __attribute__((ext_vector_type(2)))  _Float16 v2h;
typedef __attribute__((ext_vector_type(8)))  float    v8f;
typedef __attribute__((ext_vector_type(4)))  float    v4f;
typedef __attribute__((ext_vector_type(4)))  unsigned u32x4;
typedef __attribute__((ext_vector_type(8)))  int      i32x8;
typedef __attribute__((ext_vector_type(4)))  int      i32x4;

// ---------------------------------------------------------------------------
// CDNA5 helpers
// ---------------------------------------------------------------------------
__device__ __forceinline__ v8f wmma16(v16h a, v16h b, v8f c) {
  return __builtin_amdgcn_wmma_f32_16x16x32_f16(false, a, false, b, (short)0, c,
                                                false, false);
}

// Async global->LDS 16-byte copy (ASYNCcnt-tracked). Generic shared pointers
// carry the LDS byte offset in their low 32 bits.
__device__ __forceinline__ void async_copy16(void* lds_dst, const void* gsrc) {
  unsigned loff = (unsigned)(uintptr_t)lds_dst;
  unsigned long long g = (unsigned long long)(uintptr_t)gsrc;
  asm volatile("global_load_async_to_lds_b128 %0, %1, off"
               :
               : "v"(loff), "v"(g)
               : "memory");
}

__device__ __forceinline__ void async_wait0() {
  asm volatile("s_wait_asynccnt 0" ::: "memory");
}

// Tensor Data Mover: load a 2D tile (128 rows x 32 f16) of a row-major
// [M x K] f16 tensor into LDS, inserting 4 DWORDs of pad every 16 DWORDs
// (i.e., rows land at stride 40 f16 = LDT). Descriptor per ISA 08 §8:
//   g0: count=1 | lds_addr | global_addr[56:0] | type=2
//   g1: data_size=2B, pad_enable, pad_interval=3 (16DW), pad_amount=3 (4DW),
//       tensor_dim0=K, tensor_dim1=M, tile_dim0=32, tile_dim1=128,
//       tensor_dim0_stride=K (data_size units)
__device__ __forceinline__ void tdm_load_tile_a(void* lds_dst,
                                                const void* gsrc, int K,
                                                int M) {
  unsigned long long g = (unsigned long long)(uintptr_t)gsrc;
  u32x4 g0;
  g0[0] = 1u;                                    // count=1 (valid descriptor)
  g0[1] = (unsigned)(uintptr_t)lds_dst;          // lds_addr
  g0[2] = (unsigned)(g & 0xffffffffu);           // global_addr[31:0]
  g0[3] = ((unsigned)(g >> 32) & 0x01ffffffu) |  // global_addr[56:32]
          0x80000000u;                           // type=2 ("image")
  i32x8 g1;
  g1[0] = (int)((1u << 16) |   // data_size = 2 bytes
                (1u << 20) |   // pad_enable
                (3u << 22) |   // pad_interval: 16 DWORDs
                (3u << 25));   // pad_amount:   4 DWORDs
  g1[1] = (int)(((unsigned)K & 0xffffu) << 16);  // tensor_dim0 lo16 @ bits63:48
  g1[2] = (int)((((unsigned)K >> 16) & 0xffffu) |
                (((unsigned)M & 0xffffu) << 16));  // dim0 hi | dim1 lo
  g1[3] = (int)((((unsigned)M >> 16) & 0xffffu) |
                (32u << 16));                      // dim1 hi | tile_dim0=32
  g1[4] = 128;                                     // tile_dim1 (tile_dim2=0)
  g1[5] = K;                                       // tensor_dim0_stride lo32
  g1[6] = 0;
  g1[7] = 0;
  i32x4 z4 = {0, 0, 0, 0};
  i32x8 z8 = {0, 0, 0, 0, 0, 0, 0, 0};
  __builtin_amdgcn_tensor_load_to_lds(g0, g1, z4, z4, z8, 0);
}

__device__ __forceinline__ void tensor_wait0() {
  __builtin_amdgcn_s_wait_tensorcnt(0);
}

// Load a 16x32 f16 A-style operand fragment from a row-major array.
// Lane layout (CDNA5 ISA 7.12.2): lanes 0-15 hold row M=lane, K=0..7 (v0..3)
// and K=16..23 (v4..7); lanes 16-31 hold K=8..15 and K=24..31.
__device__ __forceinline__ v16h load_frag(const _Float16* base, int ldm,
                                          int row0, int kbase) {
  int lane = threadIdx.x & 31;
  const _Float16* p = base + (size_t)(row0 + (lane & 15)) * ldm + kbase +
                      ((lane >> 4) << 3);
  v8h lo = *(const v8h*)(p);
  v8h hi = *(const v8h*)(p + 16);
  v16h f;
#pragma unroll
  for (int i = 0; i < 8; ++i) { f[i] = lo[i]; f[i + 8] = hi[i]; }
  return f;
}

// ---------------------------------------------------------------------------
// GEMM: C[M,N] (f32) = A[M,K] (f16, row-major) * B[K,N] (f16, row-major)
// Block tile 128x128, BK=32, 256 threads = 8 waves, wave tile 32x64.
// A tile: TDM (tensor_load_to_lds) descriptor issued by wave 0, one k-step
// ahead into the ping-pong buffer, fenced with s_wait_tensorcnt.
// B tile: register-staged with on-the-fly transpose (ds_store_b32 pairs).
// ---------------------------------------------------------------------------
#define BM 128
#define BN 128
#define BK 32
#define LDT (BK + 8)

__global__ __launch_bounds__(256) void gemm_f16f32(
    const _Float16* __restrict__ A, const _Float16* __restrict__ B,
    float* __restrict__ C, int M, int N, int K) {
  __shared__ __align__(16) _Float16 As[2][BM][LDT];
  __shared__ __align__(16) _Float16 BsT[2][BN][LDT];

  const int t = threadIdx.x;
  const int wave = t >> 5;
  const int lane = t & 31;
  const int m0 = blockIdx.y * BM;
  const int n0 = blockIdx.x * BN;
  const int wm = (wave & 3) * 32;   // 4 wave rows * 32
  const int wn = (wave >> 2) * 64;  // 2 wave cols * 64

  // B staging: thread covers k-pair (bkp, bkp+1) x 8 n's -> ds_store_b32
  const int bkp = (t & 15) * 2, bn8 = (t >> 4) * 8;
  const _Float16* bptr = B + (size_t)bkp * N + n0 + bn8;
  const _Float16* atile = A + (size_t)m0 * K;  // tile row base (uniform)

  // prologue: TDM A tile 0, manual B tile 0
  if (wave == 0) tdm_load_tile_a(&As[0][0][0], atile, K, M);
  v8h b0 = *(const v8h*)(bptr);
  v8h b1 = *(const v8h*)(bptr + N);
#pragma unroll
  for (int j = 0; j < 8; ++j) {
    v2h p; p[0] = b0[j]; p[1] = b1[j];
    *(v2h*)&BsT[0][bn8 + j][bkp] = p;
  }
  if (wave == 0) tensor_wait0();

  v8f zero8 = {0.f, 0.f, 0.f, 0.f, 0.f, 0.f, 0.f, 0.f};
  v8f acc[2][4];
#pragma unroll
  for (int mi = 0; mi < 2; ++mi)
#pragma unroll
    for (int ni = 0; ni < 4; ++ni) acc[mi][ni] = zero8;

  const int nk = K / BK;
  for (int kt = 0; kt < nk; ++kt) {
    __syncthreads();
    const int cur = kt & 1, nxt = cur ^ 1;

    if (kt + 1 < nk) {
      // TDM: A tile kt+1 straight into the other LDS buffer (wave 0 only)
      if (wave == 0)
        tdm_load_tile_a(&As[nxt][0][0], atile + (size_t)(kt + 1) * BK, K, M);
      // B tile kt+1 into registers (stored after compute)
      const _Float16* bp = bptr + (size_t)(kt + 1) * BK * N;
      b0 = *(const v8h*)(bp);
      b1 = *(const v8h*)(bp + N);
    }
    if (kt + 2 < nk)  // L2 prefetch two steps ahead (B path)
      __builtin_prefetch(bptr + (size_t)(kt + 2) * BK * N, 0, 0);

    v16h af[2], bf[4];
    af[0] = load_frag(&As[cur][0][0], LDT, wm, 0);
    af[1] = load_frag(&As[cur][0][0], LDT, wm + 16, 0);
#pragma unroll
    for (int ni = 0; ni < 4; ++ni)
      bf[ni] = load_frag(&BsT[cur][0][0], LDT, wn + ni * 16, 0);
#pragma unroll
    for (int mi = 0; mi < 2; ++mi)
#pragma unroll
      for (int ni = 0; ni < 4; ++ni)
        acc[mi][ni] = wmma16(af[mi], bf[ni], acc[mi][ni]);

    if (kt + 1 < nk) {  // park transposed B tile into the other LDS buffer
#pragma unroll
      for (int j = 0; j < 8; ++j) {
        v2h p; p[0] = b0[j]; p[1] = b1[j];
        *(v2h*)&BsT[nxt][bn8 + j][bkp] = p;
      }
      if (wave == 0) tensor_wait0();  // TDM tile landed before next barrier
    }
  }

  const int coll = lane & 15;
  const int rsel = (lane >> 4) * 8;
#pragma unroll
  for (int mi = 0; mi < 2; ++mi) {
    float* cbase = C + (size_t)(m0 + wm + mi * 16 + rsel) * N + n0 + wn + coll;
#pragma unroll
    for (int r = 0; r < 8; ++r)
#pragma unroll
      for (int ni = 0; ni < 4; ++ni)
        cbase[(size_t)r * N + ni * 16] = acc[mi][ni][r];
  }
}

// ---------------------------------------------------------------------------
// Fused causal flash attention (GQA rep=2). One block = (head, 64 q rows),
// 4 waves, each wave 16 q rows. Online softmax, WMMA for QK^T and P*V.
// K tile staged with async global->LDS; V tile transposed via registers.
// ---------------------------------------------------------------------------
__global__ __launch_bounds__(128) void flash_attn(
    const _Float16* __restrict__ qh, const _Float16* __restrict__ kh,
    const _Float16* __restrict__ vh, _Float16* __restrict__ yh) {
  __shared__ __align__(16) _Float16 ks[32][72];      // 32 keys x HD
  __shared__ __align__(16) _Float16 vT[HD][40];      // HD x 32 keys (transposed)
  __shared__ __align__(16) _Float16 ps[4][16][40];   // per-wave P relayout

  const int t = threadIdx.x;
  const int wave = t >> 5;
  const int lane = t & 31;
  const int head = blockIdx.x;
  const int kvh = head >> 1;  // NH/NKV = 2
  const int q0 = blockIdx.y * 64 + wave * 16;

  const _Float16* qbase = qh + (size_t)q0 * (NH * HD) + head * HD;
  v16h qf0 = load_frag(qbase, NH * HD, 0, 0);
  v16h qf1 = load_frag(qbase, NH * HD, 0, 32);

  v8f zero8 = {0.f, 0.f, 0.f, 0.f, 0.f, 0.f, 0.f, 0.f};
  v8f O[4] = {zero8, zero8, zero8, zero8};
  float mrow[8], lrow[8];
#pragma unroll
  for (int r = 0; r < 8; ++r) { mrow[r] = -1e30f; lrow[r] = 0.f; }

  const int coll = lane & 15;
  const int rsel = (lane >> 4) * 8;
  const int nkb = blockIdx.y * 2 + 2;  // keys up to q-block end; mask exacts it

  for (int kb = 0; kb < nkb; ++kb) {
    const int k0 = kb * 32;
    __syncthreads();
    {  // stage K block via async copies (row-major, 16B chunks)
      int key = t >> 2, part = (t & 3) * 16;
      const _Float16* src = kh + ((size_t)(k0 + key) * NKV + kvh) * HD + part;
      async_copy16(&ks[key][part], src);
      async_copy16(&ks[key][part + 8], src + 8);
    }
    {  // stage V block transposed: key-pairs -> ds_store_b32
      int kp = (t & 15) * 2, hd8 = (t >> 4) * 8;
      const _Float16* s0p = vh + ((size_t)(k0 + kp) * NKV + kvh) * HD + hd8;
      const _Float16* s1p = s0p + (size_t)NKV * HD;
      v8h r0 = *(const v8h*)s0p;
      v8h r1 = *(const v8h*)s1p;
#pragma unroll
      for (int j = 0; j < 8; ++j) {
        v2h p; p[0] = r0[j]; p[1] = r1[j];
        *(v2h*)&vT[hd8 + j][kp] = p;
      }
    }
    async_wait0();
    __syncthreads();

    // scores: two 16x16 tiles (keys k0..15, k0+16..31), K-dim = HD = 2 WMMAs
    v8f s0 = zero8, s1 = zero8;
    s0 = wmma16(qf0, load_frag(&ks[0][0], 72, 0, 0), s0);
    s0 = wmma16(qf1, load_frag(&ks[0][0], 72, 0, 32), s0);
    s1 = wmma16(qf0, load_frag(&ks[0][0], 72, 16, 0), s1);
    s1 = wmma16(qf1, load_frag(&ks[0][0], 72, 16, 32), s1);

#pragma unroll
    for (int r = 0; r < 8; ++r) {
      const int qpos = q0 + r + rsel;
      float x0 = s0[r] * 0.125f + ((k0 + coll) <= qpos ? 0.0f : -255.0f);
      float x1 = s1[r] * 0.125f + ((k0 + 16 + coll) <= qpos ? 0.0f : -255.0f);
      float rm = fmaxf(x0, x1);  // half-wave (16-lane) row reduction
      rm = fmaxf(rm, __shfl_xor(rm, 8, 32));
      rm = fmaxf(rm, __shfl_xor(rm, 4, 32));
      rm = fmaxf(rm, __shfl_xor(rm, 2, 32));
      rm = fmaxf(rm, __shfl_xor(rm, 1, 32));
      float mnew = fmaxf(mrow[r], rm);
      float alpha = __expf(mrow[r] - mnew);
      float p0 = __expf(x0 - mnew);
      float p1 = __expf(x1 - mnew);
      float rs = p0 + p1;
      rs += __shfl_xor(rs, 8, 32);
      rs += __shfl_xor(rs, 4, 32);
      rs += __shfl_xor(rs, 2, 32);
      rs += __shfl_xor(rs, 1, 32);
      lrow[r] = lrow[r] * alpha + rs;
      mrow[r] = mnew;
#pragma unroll
      for (int ni = 0; ni < 4; ++ni) O[ni][r] *= alpha;
      // C-layout -> A-layout via per-wave LDS patch
      ps[wave][r + rsel][coll]      = (_Float16)p0;
      ps[wave][r + rsel][16 + coll] = (_Float16)p1;
    }
    v16h pf = load_frag(&ps[wave][0][0], 40, 0, 0);
#pragma unroll
    for (int ni = 0; ni < 4; ++ni)
      O[ni] = wmma16(pf, load_frag(&vT[0][0], 40, ni * 16, 0), O[ni]);
  }

#pragma unroll
  for (int r = 0; r < 8; ++r) {
    float invl = 1.0f / lrow[r];
    int row = q0 + r + rsel;
#pragma unroll
    for (int ni = 0; ni < 4; ++ni)
      yh[(size_t)row * (NH * HD) + head * HD + ni * 16 + coll] =
          (_Float16)(O[ni][r] * invl);
  }
}

// ---------------------------------------------------------------------------
// Elementwise / small kernels
// ---------------------------------------------------------------------------
__global__ __launch_bounds__(256) void embed_kernel(
    const int* __restrict__ tokens, const float* __restrict__ emb,
    float* __restrict__ h) {
  int s = blockIdx.x;
  int tok = tokens[s];
  for (int i = threadIdx.x; i < DIM; i += 256)
    h[(size_t)s * DIM + i] = emb[(size_t)tok * DIM + i];
}

__global__ __launch_bounds__(256) void rmsnorm_f16(
    const float* __restrict__ x, const float* __restrict__ w,
    _Float16* __restrict__ out) {
  __shared__ float red[256];
  int row = blockIdx.x;
  const float* xr = x + (size_t)row * DIM;
  float ss = 0.f;
  for (int i = threadIdx.x; i < DIM; i += 256) { float v = xr[i]; ss += v * v; }
  red[threadIdx.x] = ss;
  __syncthreads();
  for (int s = 128; s > 0; s >>= 1) {
    if (threadIdx.x < s) red[threadIdx.x] += red[threadIdx.x + s];
    __syncthreads();
  }
  float scale = rsqrtf(red[0] * (1.0f / DIM) + 1e-5f);
  for (int i = threadIdx.x; i < DIM; i += 256)
    out[(size_t)row * DIM + i] = (_Float16)(xr[i] * scale * w[i]);
}

__global__ __launch_bounds__(256) void rope_f16(
    const float* __restrict__ in, _Float16* __restrict__ out,
    const float* __restrict__ fc, const float* __restrict__ fs, int nheads) {
  int idx = blockIdx.x * 256 + threadIdx.x;  // S*nheads*(HD/2)
  int i = idx & 31;
  int hh = (idx >> 5) % nheads;
  int s = idx / (nheads * 32);
  if (s >= S_LEN) return;
  const float* base = in + ((size_t)s * nheads + hh) * HD;
  float xr = base[2 * i], xi = base[2 * i + 1];
  float c = fc[s * 32 + i], sn = fs[s * 32 + i];
  _Float16* ob = out + ((size_t)s * nheads + hh) * HD;
  ob[i]      = (_Float16)(xr * c - xi * sn);
  ob[32 + i] = (_Float16)(xr * sn + xi * c);
}

__global__ __launch_bounds__(256) void cvt_f32_f16_v4(
    const float* __restrict__ in, _Float16* __restrict__ out, int n4) {
  int i = blockIdx.x * 256 + threadIdx.x;
  if (i < n4) {
    v4f v = *(const v4f*)(in + (size_t)i * 4);
    v4h o;
#pragma unroll
    for (int j = 0; j < 4; ++j) o[j] = (_Float16)v[j];
    *(v4h*)(out + (size_t)i * 4) = o;
  }
}

__global__ __launch_bounds__(256) void silu_mul_f16(
    const float* __restrict__ g, const float* __restrict__ u,
    _Float16* __restrict__ out, int n) {
  int i = blockIdx.x * 256 + threadIdx.x;
  if (i < n) {
    float gv = g[i];
    float sv = gv / (1.0f + __expf(-gv));
    out[i] = (_Float16)(sv * u[i]);
  }
}

__global__ __launch_bounds__(256) void add_res(
    float* __restrict__ h, const float* __restrict__ d, int n) {
  int i = blockIdx.x * 256 + threadIdx.x;
  if (i < n) h[i] += d[i];
}

__global__ __launch_bounds__(256) void copy_last_h(
    const _Float16* __restrict__ src, float* __restrict__ dst) {
  int i = blockIdx.x * 256 + threadIdx.x;
  if (i < NKV * HD) dst[i] = (float)src[(size_t)(S_LEN - 1) * NKV * HD + i];
}

__global__ __launch_bounds__(256) void copy_last_f(
    const float* __restrict__ src, float* __restrict__ dst) {
  int i = blockIdx.x * 256 + threadIdx.x;
  if (i < NKV * HD) dst[i] = src[(size_t)(S_LEN - 1) * NKV * HD + i];
}

// ---------------------------------------------------------------------------
// Host orchestration
// ---------------------------------------------------------------------------
extern "C" void kernel_launch(void* const* d_in, const int* in_sizes, int n_in,
                              void* d_out, int out_size, void* d_ws,
                              size_t ws_size, hipStream_t stream) {
  const int*   tokens   = (const int*)d_in[0];
  // d_in[1] atten_mask unused: causal mask computed inline (same values)
  const float* fcos     = (const float*)d_in[2];
  const float* fsin     = (const float*)d_in[3];
  const float* tok_emb  = (const float*)d_in[4];
  const float* wq       = (const float*)d_in[5];
  const float* wk       = (const float*)d_in[6];
  const float* wv       = (const float*)d_in[7];
  const float* wo       = (const float*)d_in[8];
  const float* attn_nw  = (const float*)d_in[9];
  const float* ffn_nw   = (const float*)d_in[10];
  const float* w1       = (const float*)d_in[11];
  const float* w2       = (const float*)d_in[12];
  const float* w3       = (const float*)d_in[13];
  const float* final_nw = (const float*)d_in[14];
  const float* out_w    = (const float*)d_in[15];
  float* out = (float*)d_out;

  char* ws = (char*)d_ws;
  size_t off = 0;
  auto alloc = [&](size_t bytes) -> void* {
    void* p = ws + off;
    off = (off + bytes + 255) & ~(size_t)255;
    return p;
  };

  float*     h    = (float*)alloc((size_t)S_LEN * DIM * 4);
  _Float16*  xh   = (_Float16*)alloc((size_t)S_LEN * DIM * 2);
  _Float16*  wbuf = (_Float16*)alloc((size_t)DIM * VOCAB * 2);  // f16 weights
  float*     qf   = (float*)alloc((size_t)S_LEN * NH * HD * 4); // also wo/w2 out
  float*     kf   = (float*)alloc((size_t)S_LEN * NKV * HD * 4);
  float*     vf   = (float*)alloc((size_t)S_LEN * NKV * HD * 4);
  _Float16*  qh   = (_Float16*)alloc((size_t)S_LEN * NH * HD * 2);
  _Float16*  khb  = (_Float16*)alloc((size_t)S_LEN * NKV * HD * 2);
  _Float16*  vhb  = (_Float16*)alloc((size_t)S_LEN * NKV * HD * 2);
  _Float16*  yh   = (_Float16*)alloc((size_t)S_LEN * NH * HD * 2);
  float*     gbuf = (float*)alloc((size_t)S_LEN * HID * 4);
  float*     ubuf = (float*)alloc((size_t)S_LEN * HID * 4);
  _Float16*  th   = (_Float16*)alloc((size_t)S_LEN * HID * 2);

  auto gemm = [&](const _Float16* A, const float* Bw, float* C, int M, int N,
                  int K) {
    int n4 = (K * N) / 4;
    cvt_f32_f16_v4<<<(n4 + 255) / 256, 256, 0, stream>>>(Bw, wbuf, n4);
    dim3 grid(N / BN, M / BM);
    gemm_f16f32<<<grid, 256, 0, stream>>>(A, wbuf, C, M, N, K);
  };

  embed_kernel<<<S_LEN, 256, 0, stream>>>(tokens, tok_emb, h);

  for (int l = 0; l < NL; ++l) {
    rmsnorm_f16<<<S_LEN, 256, 0, stream>>>(h, attn_nw + (size_t)l * DIM, xh);
    gemm(xh, wq + (size_t)l * DIM * NH * HD,  qf, S_LEN, NH * HD,  DIM);
    gemm(xh, wk + (size_t)l * DIM * NKV * HD, kf, S_LEN, NKV * HD, DIM);
    gemm(xh, wv + (size_t)l * DIM * NKV * HD, vf, S_LEN, NKV * HD, DIM);

    rope_f16<<<(S_LEN * NH * 32 + 255) / 256, 256, 0, stream>>>(qf, qh, fcos,
                                                                fsin, NH);
    rope_f16<<<(S_LEN * NKV * 32 + 255) / 256, 256, 0, stream>>>(kf, khb, fcos,
                                                                 fsin, NKV);
    cvt_f32_f16_v4<<<(S_LEN * NKV * HD / 4 + 255) / 256, 256, 0, stream>>>(
        vf, vhb, S_LEN * NKV * HD / 4);

    flash_attn<<<dim3(NH, S_LEN / 64), 128, 0, stream>>>(qh, khb, vhb, yh);

    gemm(yh, wo + (size_t)l * NH * HD * DIM, qf, S_LEN, DIM, NH * HD);
    add_res<<<(S_LEN * DIM + 255) / 256, 256, 0, stream>>>(h, qf, S_LEN * DIM);

    rmsnorm_f16<<<S_LEN, 256, 0, stream>>>(h, ffn_nw + (size_t)l * DIM, xh);
    gemm(xh, w1 + (size_t)l * DIM * HID, gbuf, S_LEN, HID, DIM);
    gemm(xh, w3 + (size_t)l * DIM * HID, ubuf, S_LEN, HID, DIM);
    silu_mul_f16<<<(S_LEN * HID + 255) / 256, 256, 0, stream>>>(
        gbuf, ubuf, th, S_LEN * HID);
    gemm(th, w2 + (size_t)l * HID * DIM, qf, S_LEN, DIM, HID);
    add_res<<<(S_LEN * DIM + 255) / 256, 256, 0, stream>>>(h, qf, S_LEN * DIM);

    copy_last_h<<<2, 256, 0, stream>>>(
        khb, out + (size_t)S_LEN * VOCAB + (size_t)l * NKV * HD);
    copy_last_f<<<2, 256, 0, stream>>>(
        vf, out + (size_t)S_LEN * VOCAB + (size_t)NL * NKV * HD +
                (size_t)l * NKV * HD);
  }

  rmsnorm_f16<<<S_LEN, 256, 0, stream>>>(h, final_nw, xh);
  gemm(xh, out_w, out, S_LEN, VOCAB, DIM);
}
